// MDViT_57337813401610
// MI455X (gfx1250) — matrix operand order repaired and checked by reading, not compiled, not run on hardware
//
#include <hip/hip_runtime.h>
#include <math.h>

// ---------------------------------------------------------------------------
// MDViT block for MI455X (gfx1250, wave32, WMMA).
// Big GEMMs (QKV / proj / MLP1 / MLP2, ~81 GFLOP) run on v_wmma_f32_16x16x32_bf16.
// Wave tile 64x32 (4x2 wmma tiles): 8 wmma per 6 fragment loads per K-step.
// Small einsums + depthwise convs are LDS-staged f32 kernels.
// ---------------------------------------------------------------------------

typedef __bf16 v16bf __attribute__((ext_vector_type(16)));
typedef __bf16 v8bf  __attribute__((ext_vector_type(8)));
typedef float  v8f   __attribute__((ext_vector_type(8)));

#define NB   8          // batch
#define NH   8          // heads
#define CH   40         // channels / head
#define CC   320        // C
#define HH   64
#define WW   64
#define NN   4096       // tokens
#define HID  1280       // mlp hidden

// ---------------------------------------------------------------------------
// 1. ConvPosEnc: depthwise 3x3 + residual.  x:(B,N,C) -> x1:(B,N,C)
// ---------------------------------------------------------------------------
__global__ void __launch_bounds__(256) cpe_kernel(const float* __restrict__ x,
                                                  const float* __restrict__ w,
                                                  const float* __restrict__ bias,
                                                  float* __restrict__ out) {
  size_t idx = (size_t)blockIdx.x * 256 + threadIdx.x;       // < B*N*C
  int c = (int)(idx % CC);
  size_t rem = idx / CC;
  int n = (int)(rem & (NN - 1));
  int b = (int)(rem >> 12);
  int y = n >> 6, xx = n & 63;
  const float* xb = x + (size_t)b * NN * CC;
  float acc = bias[c];
  #pragma unroll
  for (int dy = 0; dy < 3; dy++) {
    int yy = y + dy - 1;
    if (yy < 0 || yy >= HH) continue;
    #pragma unroll
    for (int dx = 0; dx < 3; dx++) {
      int xc = xx + dx - 1;
      if (xc < 0 || xc >= WW) continue;
      acc += xb[((size_t)yy * WW + xc) * CC + c] * w[c * 9 + dy * 3 + dx];
    }
  }
  out[idx] = acc + xb[(size_t)n * CC + c];
}

// ---------------------------------------------------------------------------
// 2. LayerNorm over C, emits bf16 for the WMMA GEMMs.  one block / row
// ---------------------------------------------------------------------------
__global__ void __launch_bounds__(128) ln_kernel(const float* __restrict__ x,
                                                 const float* __restrict__ g,
                                                 const float* __restrict__ b,
                                                 __bf16* __restrict__ out) {
  __shared__ float s1[128], s2[128];
  const size_t row = blockIdx.x;
  const float* xp = x + row * CC;
  int t = threadIdx.x;
  float s = 0.f, ss = 0.f;
  for (int c = t; c < CC; c += 128) { float v = xp[c]; s += v; ss += v * v; }
  s1[t] = s; s2[t] = ss; __syncthreads();
  for (int st = 64; st > 0; st >>= 1) {
    if (t < st) { s1[t] += s1[t + st]; s2[t] += s2[t + st]; }
    __syncthreads();
  }
  float mean = s1[0] * (1.f / CC);
  float var  = s2[0] * (1.f / CC) - mean * mean;
  float inv  = rsqrtf(var + 1e-6f);
  __bf16* op = out + row * CC;
  for (int c = t; c < CC; c += 128)
    op[c] = (__bf16)((xp[c] - mean) * inv * g[c] + b[c]);
}

// ---------------------------------------------------------------------------
// 3. f32 -> bf16 weight conversion
// ---------------------------------------------------------------------------
__global__ void __launch_bounds__(256) cvt_bf16(const float* __restrict__ in,
                                                __bf16* __restrict__ out, int n) {
  int i = blockIdx.x * 256 + threadIdx.x;
  if (i < n) out[i] = (__bf16)in[i];
}

// ---------------------------------------------------------------------------
// 4. WMMA bf16 GEMM:  out[m,n] = sum_k A[m,k] * Wt[n,k]  (+bias, fused epilogue)
//    block = 128 threads = 4 waves (2M x 2N); block tile 128x64
//    wave tile 64x32 = 4x2 wmma tiles -> 8 wmma / 6 fragment loads per K-step
//    mode 0: scatter qkv -> q/k/v (B,h,N,Ch) f32
//    mode 1: f32 out + residual
//    mode 2: bf16 out with exact GELU
// ---------------------------------------------------------------------------
__device__ __forceinline__ v16bf load_a_frag(const __bf16* p) {
  // A fragment in this code's contiguous k order (B matches): lane<16 -> K {0..15}; lane>=16 -> K {16..31}
  return *reinterpret_cast<const v16bf*>(p);
}

__device__ __forceinline__ float gelu_exact(float x) {
  return 0.5f * x * (1.f + erff(x * 0.70710678118654752f));
}

__global__ void __launch_bounds__(128) gemm_wmma_bf16(
    const __bf16* __restrict__ A, const __bf16* __restrict__ Wt,
    const float* __restrict__ bias, int K, int N, int mode,
    float* __restrict__ out_f32, const float* __restrict__ residual,
    __bf16* __restrict__ out_bf16,
    float* __restrict__ qp, float* __restrict__ kp, float* __restrict__ vp) {
  const int lane  = threadIdx.x & 31;
  const int wave  = threadIdx.x >> 5;     // 0..3
  const int waveM = wave & 1;             // 2 waves in M
  const int waveN = wave >> 1;            // 2 waves in N
  const int row0  = blockIdx.x * 128 + waveM * 64;
  const int col0  = blockIdx.y * 64  + waveN * 32;
  const int hi    = lane >> 4;
  const int lo    = lane & 15;

  const __bf16* Ar[4];
  #pragma unroll
  for (int ti = 0; ti < 4; ti++)
    Ar[ti] = A + (size_t)(row0 + ti * 16 + lo) * K + hi * 16;
  const __bf16* Wr0 = Wt + (size_t)(col0 + lo)      * K + hi * 16;
  const __bf16* Wr1 = Wt + (size_t)(col0 + 16 + lo) * K + hi * 16;

  v8f acc[4][2] = {};

  for (int k = 0; k < K; k += 32) {
    v16bf b0 = *reinterpret_cast<const v16bf*>(Wr0 + k);
    v16bf b1 = *reinterpret_cast<const v16bf*>(Wr1 + k);
    v16bf a0 = load_a_frag(Ar[0] + k);
    v16bf a1 = load_a_frag(Ar[1] + k);
    v16bf a2 = load_a_frag(Ar[2] + k);
    v16bf a3 = load_a_frag(Ar[3] + k);
    acc[0][0] = __builtin_amdgcn_wmma_f32_16x16x32_bf16(false, a0, false, b0, (short)0, acc[0][0], false, false);
    acc[0][1] = __builtin_amdgcn_wmma_f32_16x16x32_bf16(false, a0, false, b1, (short)0, acc[0][1], false, false);
    acc[1][0] = __builtin_amdgcn_wmma_f32_16x16x32_bf16(false, a1, false, b0, (short)0, acc[1][0], false, false);
    acc[1][1] = __builtin_amdgcn_wmma_f32_16x16x32_bf16(false, a1, false, b1, (short)0, acc[1][1], false, false);
    acc[2][0] = __builtin_amdgcn_wmma_f32_16x16x32_bf16(false, a2, false, b0, (short)0, acc[2][0], false, false);
    acc[2][1] = __builtin_amdgcn_wmma_f32_16x16x32_bf16(false, a2, false, b1, (short)0, acc[2][1], false, false);
    acc[3][0] = __builtin_amdgcn_wmma_f32_16x16x32_bf16(false, a3, false, b0, (short)0, acc[3][0], false, false);
    acc[3][1] = __builtin_amdgcn_wmma_f32_16x16x32_bf16(false, a3, false, b1, (short)0, acc[3][1], false, false);
    // hide next K-tile latency (emits global_prefetch_b8)
    __builtin_prefetch(Ar[0] + k + 32, 0, 1);
    __builtin_prefetch(Ar[1] + k + 32, 0, 1);
    __builtin_prefetch(Ar[2] + k + 32, 0, 1);
    __builtin_prefetch(Ar[3] + k + 32, 0, 1);
    __builtin_prefetch(Wr0 + k + 32, 0, 1);
    __builtin_prefetch(Wr1 + k + 32, 0, 1);
  }

  #pragma unroll
  for (int ti = 0; ti < 4; ti++) {
    #pragma unroll
    for (int tj = 0; tj < 2; tj++) {
      // C/D layout: lane holds n = lo, m = hi*8 + r  (r = vgpr index)
      int mBase = row0 + ti * 16 + hi * 8;
      int nIdx  = col0 + tj * 16 + lo;
      float bv = bias[nIdx];
      #pragma unroll
      for (int r = 0; r < 8; r++) {
        int m = mBase + r;
        float val = acc[ti][tj][r] + bv;
        if (mode == 0) {                      // qkv scatter
          int which = nIdx / CC;
          int c  = nIdx % CC;
          int hd = c / CH, ch = c % CH;
          int b  = m >> 12, tok = m & (NN - 1);
          float* dp = (which == 0) ? qp : (which == 1) ? kp : vp;
          dp[((size_t)(b * NH + hd) * NN + tok) * CH + ch] = val;
        } else if (mode == 1) {               // f32 + residual
          size_t o = (size_t)m * N + nIdx;
          out_f32[o] = val + residual[o];
        } else {                              // bf16 GELU
          out_bf16[(size_t)m * N + nIdx] = (__bf16)gelu_exact(val);
        }
      }
    }
  }
}

// ---------------------------------------------------------------------------
// 5. softmax over tokens for each (b,h,ch) column of k.  LDS-resident column.
// ---------------------------------------------------------------------------
__global__ void __launch_bounds__(256) softmax_col(float* __restrict__ kbuf) {
  __shared__ float buf[NN];
  __shared__ float red[256];
  int col = blockIdx.x;               // 0..B*h*Ch-1
  int bh = col / CH, ch = col % CH;
  float* base = kbuf + (size_t)bh * NN * CH + ch;
  int t = threadIdx.x;
  float lmax = -1e30f;
  for (int n = t; n < NN; n += 256) { float v = base[(size_t)n * CH]; buf[n] = v; lmax = fmaxf(lmax, v); }
  red[t] = lmax; __syncthreads();
  for (int st = 128; st > 0; st >>= 1) { if (t < st) red[t] = fmaxf(red[t], red[t + st]); __syncthreads(); }
  float mx = red[0]; __syncthreads();
  float lsum = 0.f;
  for (int n = t; n < NN; n += 256) { float e = __expf(buf[n] - mx); buf[n] = e; lsum += e; }
  red[t] = lsum; __syncthreads();
  for (int st = 128; st > 0; st >>= 1) { if (t < st) red[t] += red[t + st]; __syncthreads(); }
  float inv = 1.f / red[0];
  for (int n = t; n < NN; n += 256) base[(size_t)n * CH] = buf[n] * inv;
}

// ---------------------------------------------------------------------------
// 6. kv = softmax(k)^T @ v per (b,h): (40x4096)x(4096x40), LDS-tiled.
// ---------------------------------------------------------------------------
__global__ void __launch_bounds__(256) kv_kernel(const float* __restrict__ ksm,
                                                 const float* __restrict__ v,
                                                 float* __restrict__ kv) {
  __shared__ float tk[128 * CH];
  __shared__ float tv[128 * CH];
  int bh = blockIdx.x;
  const float* kb = ksm + (size_t)bh * NN * CH;
  const float* vb = v   + (size_t)bh * NN * CH;
  float acc[7] = {0.f, 0.f, 0.f, 0.f, 0.f, 0.f, 0.f};
  for (int n0 = 0; n0 < NN; n0 += 128) {
    __syncthreads();
    for (int i = threadIdx.x; i < 128 * CH; i += 256) {
      tk[i] = kb[(size_t)n0 * CH + i];
      tv[i] = vb[(size_t)n0 * CH + i];
    }
    __syncthreads();
    #pragma unroll
    for (int pi = 0; pi < 7; pi++) {
      int p = threadIdx.x + pi * 256;
      if (p < CH * CH) {
        int ck = p / CH, cv = p % CH;
        float a = 0.f;
        for (int r = 0; r < 128; r++) a += tk[r * CH + ck] * tv[r * CH + cv];
        acc[pi] += a;
      }
    }
  }
  #pragma unroll
  for (int pi = 0; pi < 7; pi++) {
    int p = threadIdx.x + pi * 256;
    if (p < CH * CH) kv[(size_t)bh * CH * CH + p] = acc[pi];
  }
}

// ---------------------------------------------------------------------------
// 7. domain gate: softmax-over-heads of MLP(one_hot).  one block per batch.
// ---------------------------------------------------------------------------
__global__ void __launch_bounds__(128) gate_kernel(const float* __restrict__ dl,
                                                   const float* __restrict__ w1, const float* __restrict__ b1,
                                                   const float* __restrict__ w2, const float* __restrict__ b2,
                                                   float* __restrict__ gate) {
  __shared__ float d1[80];
  __shared__ float d2[CC];
  int b = blockIdx.x, t = threadIdx.x;
  const float* lab = dl + b * 4;
  if (t < 80) {
    float s = b1[t];
    #pragma unroll
    for (int j = 0; j < 4; j++) s += w1[t * 4 + j] * lab[j];
    d1[t] = fmaxf(s, 0.f);
  }
  __syncthreads();
  for (int i = t; i < CC; i += 128) {
    float s = b2[i];
    for (int j = 0; j < 80; j++) s += w2[i * 80 + j] * d1[j];
    d2[i] = s;
  }
  __syncthreads();
  if (t < CH) {
    float mx = -1e30f;
    #pragma unroll
    for (int hd = 0; hd < NH; hd++) mx = fmaxf(mx, d2[hd * CH + t]);
    float e[NH], sum = 0.f;
    #pragma unroll
    for (int hd = 0; hd < NH; hd++) { e[hd] = __expf(d2[hd * CH + t] - mx); sum += e[hd]; }
    float inv = 1.f / sum;
    #pragma unroll
    for (int hd = 0; hd < NH; hd++) gate[((size_t)b * NH + hd) * CH + t] = e[hd] * inv;
  }
}

// ---------------------------------------------------------------------------
// 8. att = gate * (scale * q@kv + q * dwconv(v))  ->  bf16 (B,N,C)
//    one block per (b,head,y) image row; kv + q-row staged in LDS.
// ---------------------------------------------------------------------------
__global__ void __launch_bounds__(256) att_kernel(const float* __restrict__ q,
                                                  const float* __restrict__ v,
                                                  const float* __restrict__ kv,
                                                  const float* __restrict__ gate,
                                                  const float* __restrict__ w3, const float* __restrict__ b3,
                                                  const float* __restrict__ w5, const float* __restrict__ b5,
                                                  const float* __restrict__ w7, const float* __restrict__ b7,
                                                  __bf16* __restrict__ att) {
  __shared__ float kvS[CH * CH];
  __shared__ float qS[WW * CH];
  int blk = blockIdx.x;               // bh*64 + y
  int y  = blk & 63;
  int bh = blk >> 6;
  int b  = bh >> 3, head = bh & 7;
  const float* qb = q + ((size_t)bh * NN + (size_t)y * WW) * CH;
  for (int i = threadIdx.x; i < CH * CH; i += 256) kvS[i] = kv[(size_t)bh * CH * CH + i];
  for (int i = threadIdx.x; i < WW * CH; i += 256) qS[i] = qb[i];
  __syncthreads();

  const float* wsel; const float* bsel; int ksz, chanBase;
  if (head < 2)      { wsel = w3; bsel = b3; ksz = 3; chanBase = head * CH; }
  else if (head < 5) { wsel = w5; bsel = b5; ksz = 5; chanBase = (head - 2) * CH; }
  else               { wsel = w7; bsel = b7; ksz = 7; chanBase = (head - 5) * CH; }
  int half = ksz >> 1;
  const float* vb = v + (size_t)bh * NN * CH;
  const float scale = 0.15811388300841897f;   // 40^-0.5

  for (int p = threadIdx.x; p < WW * CH; p += 256) {
    int x = p / CH, ch = p % CH;
    float dot = 0.f;
    for (int kk = 0; kk < CH; kk++) dot += qS[x * CH + kk] * kvS[kk * CH + ch];
    const float* wp = wsel + (size_t)(chanBase + ch) * ksz * ksz;
    float cvv = bsel[chanBase + ch];
    for (int dy = 0; dy < ksz; dy++) {
      int yy = y + dy - half;
      if (yy < 0 || yy >= HH) continue;
      for (int dx = 0; dx < ksz; dx++) {
        int xx = x + dx - half;
        if (xx < 0 || xx >= WW) continue;
        cvv += vb[((size_t)yy * WW + xx) * CH + ch] * wp[dy * ksz + dx];
      }
    }
    float g = gate[(size_t)bh * CH + ch];
    float val = g * (scale * dot + qS[x * CH + ch] * cvv);
    att[((size_t)(b * NN + y * WW + x)) * CC + head * CH + ch] = (__bf16)val;
  }
}

// ---------------------------------------------------------------------------
// host launcher
// ---------------------------------------------------------------------------
extern "C" void kernel_launch(void* const* d_in, const int* in_sizes, int n_in,
                              void* d_out, int out_size, void* d_ws, size_t ws_size,
                              hipStream_t stream) {
  const float* x        = (const float*)d_in[0];
  const float* dlabel   = (const float*)d_in[1];
  const float* cpe_w    = (const float*)d_in[2];
  const float* cpe_b    = (const float*)d_in[3];
  const float* ln1_g    = (const float*)d_in[4];
  const float* ln1_b    = (const float*)d_in[5];
  const float* qkv_w    = (const float*)d_in[6];
  const float* qkv_b    = (const float*)d_in[7];
  const float* crpe_w3  = (const float*)d_in[8];
  const float* crpe_b3  = (const float*)d_in[9];
  const float* crpe_w5  = (const float*)d_in[10];
  const float* crpe_b5  = (const float*)d_in[11];
  const float* crpe_w7  = (const float*)d_in[12];
  const float* crpe_b7  = (const float*)d_in[13];
  const float* dom_w1   = (const float*)d_in[14];
  const float* dom_b1   = (const float*)d_in[15];
  const float* dom_w2   = (const float*)d_in[16];
  const float* dom_b2   = (const float*)d_in[17];
  const float* proj_w   = (const float*)d_in[18];
  const float* proj_b   = (const float*)d_in[19];
  const float* ln2_g    = (const float*)d_in[20];
  const float* ln2_b    = (const float*)d_in[21];
  const float* mlp_w1   = (const float*)d_in[22];
  const float* mlp_b1   = (const float*)d_in[23];
  const float* mlp_w2   = (const float*)d_in[24];
  const float* mlp_b2   = (const float*)d_in[25];

  const size_t BNC = (size_t)NB * NN * CC;      // 10,485,760

  // workspace layout (bytes)
  char* ws = (char*)d_ws;
  const size_t OFF_X1   = 0;                               // f32 BNC (x after CPE; later x2 in place)
  const size_t OFF_CUR  = OFF_X1  + BNC * 4;               // bf16 BNC (LN1 out; later LN2 out)
  const size_t OFF_Q    = OFF_CUR + BNC * 2;               // f32 BNC
  const size_t OFF_K    = OFF_Q   + BNC * 4;               // f32 BNC
  const size_t OFF_V    = OFF_K   + BNC * 4;               // f32 BNC
  const size_t OFF_KV   = OFF_V   + BNC * 4;               // f32 B*h*40*40
  const size_t OFF_GATE = OFF_KV  + (size_t)NB * NH * CH * CH * 4;
  const size_t OFF_ATT  = OFF_GATE + (size_t)NB * NH * CH * 4;   // bf16 BNC
  const size_t OFF_WB   = OFF_ATT + BNC * 2;               // bf16 weights
  const size_t OFF_HIDB = OFF_Q;                           // bf16 B*N*1280, overlays dead Q/K

  float*  X1   = (float*)(ws + OFF_X1);
  __bf16* CUR  = (__bf16*)(ws + OFF_CUR);
  float*  Q    = (float*)(ws + OFF_Q);
  float*  Kb   = (float*)(ws + OFF_K);
  float*  V    = (float*)(ws + OFF_V);
  float*  KV   = (float*)(ws + OFF_KV);
  float*  GATE = (float*)(ws + OFF_GATE);
  __bf16* ATT  = (__bf16*)(ws + OFF_ATT);
  __bf16* HIDB = (__bf16*)(ws + OFF_HIDB);
  __bf16* WQKV = (__bf16*)(ws + OFF_WB);
  __bf16* WPRJ = WQKV + (size_t)3 * CC * CC;               // 960*320
  __bf16* WM1  = WPRJ + (size_t)CC * CC;                   // 320*320
  __bf16* WM2  = WM1  + (size_t)HID * CC;                  // 1280*320

  // 1. ConvPosEnc + residual
  cpe_kernel<<<(unsigned)(BNC / 256), 256, 0, stream>>>(x, cpe_w, cpe_b, X1);
  // 2. LN1 -> bf16
  ln_kernel<<<NB * NN, 128, 0, stream>>>(X1, ln1_g, ln1_b, CUR);
  // 3. weight conversions
  cvt_bf16<<<(3 * CC * CC) / 256, 256, 0, stream>>>(qkv_w, WQKV, 3 * CC * CC);
  cvt_bf16<<<(CC * CC) / 256, 256, 0, stream>>>(proj_w, WPRJ, CC * CC);
  cvt_bf16<<<(HID * CC) / 256, 256, 0, stream>>>(mlp_w1, WM1, HID * CC);
  cvt_bf16<<<(CC * HID) / 256, 256, 0, stream>>>(mlp_w2, WM2, CC * HID);
  // 4. QKV GEMM (32768x320 @ 320x960), scatter into q/k/v
  {
    dim3 g(NB * NN / 128, (3 * CC) / 64);
    gemm_wmma_bf16<<<g, 128, 0, stream>>>(CUR, WQKV, qkv_b, CC, 3 * CC, 0,
                                          nullptr, nullptr, nullptr, Q, Kb, V);
  }
  // 5. softmax(k) over tokens (in place)
  softmax_col<<<NB * NH * CH, 256, 0, stream>>>(Kb);
  // 6. kv einsum
  kv_kernel<<<NB * NH, 256, 0, stream>>>(Kb, V, KV);
  // 7. domain gate
  gate_kernel<<<NB, 128, 0, stream>>>(dlabel, dom_w1, dom_b1, dom_w2, dom_b2, GATE);
  // 8. fused factor-att + CRPE + gate -> att (bf16, B,N,C)
  att_kernel<<<NB * NH * HH, 256, 0, stream>>>(Q, V, KV, GATE,
                                               crpe_w3, crpe_b3, crpe_w5, crpe_b5,
                                               crpe_w7, crpe_b7, ATT);
  // 9. proj GEMM + residual (in place over X1 -> x2)
  {
    dim3 g(NB * NN / 128, CC / 64);
    gemm_wmma_bf16<<<g, 128, 0, stream>>>(ATT, WPRJ, proj_b, CC, CC, 1,
                                          X1, X1, nullptr, nullptr, nullptr, nullptr);
  }
  // 10. LN2 -> bf16
  ln_kernel<<<NB * NN, 128, 0, stream>>>(X1, ln2_g, ln2_b, CUR);
  // 11. MLP1 GEMM + exact GELU -> bf16 hidden
  {
    dim3 g(NB * NN / 128, HID / 64);
    gemm_wmma_bf16<<<g, 128, 0, stream>>>(CUR, WM1, mlp_b1, CC, HID, 2,
                                          nullptr, nullptr, HIDB, nullptr, nullptr, nullptr);
  }
  // 12. MLP2 GEMM + residual -> d_out
  {
    dim3 g(NB * NN / 128, CC / 64);
    gemm_wmma_bf16<<<g, 128, 0, stream>>>(HIDB, WM2, mlp_b2, HID, CC, 1,
                                          (float*)d_out, X1, nullptr, nullptr, nullptr, nullptr);
  }
  (void)in_sizes; (void)n_in; (void)out_size; (void)ws_size;
}
